// TemporalSelfAttention_17995912970832
// MI455X (gfx1250) — matrix operand order repaired and checked by reading, not compiled, not run on hardware
//
#include <hip/hip_runtime.h>
#include <math.h>

#define B_   32
#define T_   512
#define D_   512
#define H_   8
#define HD_  64
#define TED_ 2048
#define BT_  (B_ * T_)

typedef _Float16 v16h __attribute__((ext_vector_type(16)));
typedef _Float16 v8h  __attribute__((ext_vector_type(8)));
typedef float    v8f  __attribute__((ext_vector_type(8)));
typedef float    v4f  __attribute__((ext_vector_type(4)));

// ---------------- WMMA fragment helpers (gfx1250, wave32) ----------------
// A (16x32 f16, row-major source, leading dim ld in halves):
//   lane: m = lane&15, h = lane>>4 ; VGPR0-3: K=h*8..h*8+7 ; VGPR4-7: K=16+h*8..+7
__device__ __forceinline__ v16h frag_a_f16(const _Float16* __restrict__ base, int ld) {
  int lane = threadIdx.x & 31;
  int m = lane & 15, hh = lane >> 4;
  const _Float16* p = base + (size_t)m * ld + hh * 8;
  v8h lo = *(const v8h*)p;
  v8h hi = *(const v8h*)(p + 16);
  v16h r;
#pragma unroll
  for (int i = 0; i < 8; ++i) { r[i] = lo[i]; r[8 + i] = hi[i]; }
  return r;
}

// A fragment sourced from f32 data (convert on the fly)
__device__ __forceinline__ v16h frag_a_f32(const float* __restrict__ base, int ld) {
  int lane = threadIdx.x & 31;
  int m = lane & 15, hh = lane >> 4;
  const float* p = base + (size_t)m * ld + hh * 8;
  v4f a0 = *(const v4f*)p;
  v4f a1 = *(const v4f*)(p + 4);
  v4f a2 = *(const v4f*)(p + 16);
  v4f a3 = *(const v4f*)(p + 20);
  v16h r;
#pragma unroll
  for (int i = 0; i < 4; ++i) {
    r[i]      = (_Float16)a0[i];
    r[4 + i]  = (_Float16)a1[i];
    r[8 + i]  = (_Float16)a2[i];
    r[12 + i] = (_Float16)a3[i];
  }
  return r;
}

// B (32x16 f16) loaded from B^T row-major (N,K), leading dim ld:
//   lane: n = lane&15, h = lane>>4 ; VGPR0-3: K=h*8..h*8+7 ; VGPR4-7: K=16+h*8..+7
__device__ __forceinline__ v16h frag_b_f16(const _Float16* __restrict__ baseT, int ld) {
  int lane = threadIdx.x & 31;
  int n = lane & 15, hh = lane >> 4;
  const _Float16* p = baseT + (size_t)n * ld + hh * 8;
  v8h lo = *(const v8h*)p;
  v8h hi = *(const v8h*)(p + 16);
  v16h r;
#pragma unroll
  for (int i = 0; i < 8; ++i) { r[i] = lo[i]; r[8 + i] = hi[i]; }
  return r;
}

__device__ __forceinline__ v8f wmma16(v16h a, v16h b, v8f c) {
  return __builtin_amdgcn_wmma_f32_16x16x32_f16(false, a, false, b, (short)0, c, false, false);
}

// ---------------- K1: LayerNorm1 (f16 out) + sigma = f(x @ Wsig) ----------------
__global__ void k_ln1_sigma(const float* __restrict__ x, const float* __restrict__ g,
                            const float* __restrict__ bb, const float* __restrict__ Wsig,
                            _Float16* __restrict__ xn16, float* __restrict__ sig) {
  int row = blockIdx.x;           // b*T + t
  int t = threadIdx.x;            // 256 threads
  __shared__ float xr[D_];
  __shared__ float red[16];
  float x0 = x[(size_t)row * D_ + t];
  float x1 = x[(size_t)row * D_ + t + 256];
  xr[t] = x0; xr[t + 256] = x1;
  float s = x0 + x1, s2 = x0 * x0 + x1 * x1;
#pragma unroll
  for (int off = 16; off; off >>= 1) {
    s  += __shfl_xor(s, off, 32);
    s2 += __shfl_xor(s2, off, 32);
  }
  int w = t >> 5, lane = t & 31;
  if (lane == 0) { red[w] = s; red[8 + w] = s2; }
  __syncthreads();
  float ts = 0.f, ts2 = 0.f;
#pragma unroll
  for (int i = 0; i < 8; ++i) { ts += red[i]; ts2 += red[8 + i]; }
  float mean = ts * (1.f / D_);
  float var  = ts2 * (1.f / D_) - mean * mean;
  float inv  = rsqrtf(var + 1e-5f);
  xn16[(size_t)row * D_ + t]       = (_Float16)((x0 - mean) * inv * g[t] + bb[t]);
  xn16[(size_t)row * D_ + t + 256] = (_Float16)((x1 - mean) * inv * g[t + 256] + bb[t + 256]);
  // sigma: wave w computes head w
  float acc = 0.f;
  for (int i = lane; i < D_; i += 32) acc += xr[i] * Wsig[(size_t)i * H_ + w];
#pragma unroll
  for (int off = 16; off; off >>= 1) acc += __shfl_xor(acc, off, 32);
  if (lane == 0) {
    float sg = 1.f / (1.f + __expf(-5.f * acc)) + 1e-5f;
    sg = __expf(sg * 1.0986122886681098f) - 1.f;   // 3^sg - 1
    int b = row / T_, tt = row % T_;
    sig[((size_t)b * H_ + w) * T_ + tt] = sg;
  }
}

// ---------------- K2: f32 -> f16 convert ----------------
__global__ void k_cvt16(const float* __restrict__ src, _Float16* __restrict__ dst, int n) {
  int i = blockIdx.x * 256 + threadIdx.x;
  if (i < n) dst[i] = (_Float16)src[i];
}

// ---------------- K3/K9: M=16384,N=512,K=512 WMMA GEMM  C = A @ B^T ----------------
// Block tile 128x128, 8 waves (4x2), wave tile 32x64 (2x4 frags): 8 WMMA / 12 b128.
// mode 0: store f16 (M,N);  mode 2: scatter to v^T (B,H,HD,T) f16;
// mode 3: f32 out = x + C + bias  (non-temporal stores)
__global__ void k_gemm(const _Float16* __restrict__ A, const _Float16* __restrict__ Bt,
                       _Float16* __restrict__ out16, float* __restrict__ out32,
                       const float* __restrict__ xres, const float* __restrict__ bias,
                       int mode) {
  int w  = threadIdx.x >> 5;
  int wm = w >> 1, wn = w & 1;                 // 4 x 2 wave grid
  int r0 = blockIdx.x * 128 + wm * 32;
  int c0 = blockIdx.y * 128 + wn * 64;
  v8f acc[2][4] = {};
  for (int k = 0; k < D_; k += 32) {
    v16h a0 = frag_a_f16(A + (size_t)r0 * D_ + k, D_);
    v16h a1 = frag_a_f16(A + (size_t)(r0 + 16) * D_ + k, D_);
#pragma unroll
    for (int j = 0; j < 4; ++j) {
      v16h bj = frag_b_f16(Bt + (size_t)(c0 + 16 * j) * D_ + k, D_);
      acc[0][j] = wmma16(a0, bj, acc[0][j]);
      acc[1][j] = wmma16(a1, bj, acc[1][j]);
    }
  }
  int lane = threadIdx.x & 31;
  int n = lane & 15, hh = lane >> 4;
#pragma unroll
  for (int i = 0; i < 2; ++i) {
#pragma unroll
    for (int j = 0; j < 4; ++j) {
      int rr = r0 + 16 * i + hh * 8;
      int cc = c0 + 16 * j + n;
      if (mode == 0) {
#pragma unroll
        for (int v = 0; v < 8; ++v)
          out16[(size_t)(rr + v) * D_ + cc] = (_Float16)acc[i][j][v];
      } else if (mode == 2) {
        int b = rr / T_, t0 = rr % T_;       // 8 accum rows are consecutive t
        int h = cc >> 6, d = cc & 63;
        v8h pk;
#pragma unroll
        for (int v = 0; v < 8; ++v) pk[v] = (_Float16)acc[i][j][v];
        *(v8h*)(out16 + ((size_t)(b * H_ + h) * HD_ + d) * T_ + t0) = pk;
      } else {
        float bi = bias[cc];
#pragma unroll
        for (int v = 0; v < 8; ++v) {
          size_t o = (size_t)(rr + v) * D_ + cc;
          float xv = __builtin_nontemporal_load(&xres[o]);
          __builtin_nontemporal_store(xv + acc[i][j][v] + bi, &out32[o]);
        }
      }
    }
  }
}

// ---------------- K4: scores (q k^T / 64) + row softmax -> series ----------------
#define LDSN 516
__global__ void k_attn_scores(const _Float16* __restrict__ q, const _Float16* __restrict__ kk,
                              float* __restrict__ series) {
  __shared__ float sm[16 * LDSN];
  int w = threadIdx.x >> 5, lane = threadIdx.x & 31;
  int n0 = blockIdx.x * 16;
  int bh = blockIdx.y;                        // b*H + h
  int b = bh >> 3, h = bh & 7;
  const _Float16* qb = q  + ((size_t)(b * T_ + n0)) * D_ + h * HD_;
  const _Float16* kb = kk + ((size_t)b * T_) * D_ + h * HD_;
  int c0 = w * 64;
  v8f acc[4] = {};
  for (int ks = 0; ks < HD_; ks += 32) {
    v16h a = frag_a_f16(qb + ks, D_);
#pragma unroll
    for (int f = 0; f < 4; ++f) {
      v16h bf = frag_b_f16(kb + (size_t)(c0 + 16 * f) * D_ + ks, D_);
      acc[f] = wmma16(a, bf, acc[f]);
    }
  }
  int n = lane & 15, hh = lane >> 4;
#pragma unroll
  for (int f = 0; f < 4; ++f)
#pragma unroll
    for (int v = 0; v < 8; ++v)
      sm[(v + hh * 8) * LDSN + c0 + 16 * f + n] = acc[f][v] * (1.f / 64.f);
  __syncthreads();
  float* gout = series + ((size_t)bh * T_ + n0) * T_;
  for (int rr = 2 * w; rr < 2 * w + 2; ++rr) {
    float vals[16], mx = -3.4e38f;
#pragma unroll
    for (int i = 0; i < 16; ++i) {
      vals[i] = sm[rr * LDSN + lane + 32 * i];
      mx = fmaxf(mx, vals[i]);
    }
#pragma unroll
    for (int off = 16; off; off >>= 1) mx = fmaxf(mx, __shfl_xor(mx, off, 32));
    float sum = 0.f;
#pragma unroll
    for (int i = 0; i < 16; ++i) { vals[i] = __expf(vals[i] - mx); sum += vals[i]; }
#pragma unroll
    for (int off = 16; off; off >>= 1) sum += __shfl_xor(sum, off, 32);
    float invs = 1.f / sum;
#pragma unroll
    for (int i = 0; i < 16; ++i)
      gout[(size_t)rr * T_ + lane + 32 * i] = vals[i] * invs;
  }
}

// ---------------- K5: prior + sigma_full (streaming, float4, non-temporal) ----------------
__global__ void k_prior(const float* __restrict__ sig, float* __restrict__ prior,
                        float* __restrict__ sfull) {
  size_t i4 = (size_t)blockIdx.x * 256 + threadIdx.x;  // < B*H*T*T/4
  int m4 = (int)(i4 & 127);
  size_t rest = i4 >> 7;
  int nn = (int)(rest & 511);
  int bh = (int)(rest >> 9);
  float s = sig[(size_t)bh * T_ + nn];
  float inv2 = -0.5f / (s * s);
  float coef = 0.3989422804014327f / s;
  v4f p, sf;
#pragma unroll
  for (int j = 0; j < 4; ++j) {
    float d = (float)(nn - (m4 * 4 + j));
    p[j] = coef * __expf(d * d * inv2);
    sf[j] = s;
  }
  __builtin_nontemporal_store(p,  (v4f*)(prior + i4 * 4));
  __builtin_nontemporal_store(sf, (v4f*)(sfull + i4 * 4));
}

// ---------------- K6: y = series @ v  (A from f32, B from v^T f16) ----------------
__global__ void k_yv(const float* __restrict__ series, const _Float16* __restrict__ vT,
                     float* __restrict__ ymid) {
  int w  = threadIdx.x >> 5;
  int wm = w >> 1, wn = w & 1;
  int bh = blockIdx.y;
  int b = bh >> 3, h = bh & 7;
  int r0 = blockIdx.x * 128 + wm * 32;    // token rows
  int c0 = wn * 32;                       // d within head
  const float*    Ab = series + (size_t)bh * T_ * T_;
  const _Float16* Bb = vT + (size_t)bh * HD_ * T_;
  v8f acc[2][2] = {};
  for (int k = 0; k < T_; k += 32) {
    v16h a0 = frag_a_f32(Ab + (size_t)r0 * T_ + k, T_);
    v16h a1 = frag_a_f32(Ab + (size_t)(r0 + 16) * T_ + k, T_);
    v16h b0 = frag_b_f16(Bb + (size_t)c0 * T_ + k, T_);
    v16h b1 = frag_b_f16(Bb + (size_t)(c0 + 16) * T_ + k, T_);
    acc[0][0] = wmma16(a0, b0, acc[0][0]);
    acc[0][1] = wmma16(a0, b1, acc[0][1]);
    acc[1][0] = wmma16(a1, b0, acc[1][0]);
    acc[1][1] = wmma16(a1, b1, acc[1][1]);
  }
  int lane = threadIdx.x & 31;
  int n = lane & 15, hh = lane >> 4;
#pragma unroll
  for (int i = 0; i < 2; ++i)
#pragma unroll
    for (int j = 0; j < 2; ++j) {
      int rr = r0 + 16 * i + hh * 8;
      int cc = c0 + 16 * j + n;
#pragma unroll
      for (int v = 0; v < 8; ++v)
        ymid[((size_t)(b * T_ + rr + v)) * D_ + h * HD_ + cc] = acc[i][j][v];
    }
}

// ---------------- K7: emb_out = silu(emb) @ emb_W + emb_b ----------------
__global__ void k_emb(const float* __restrict__ emb, const float* __restrict__ W,
                      const float* __restrict__ bias, float* __restrict__ out) {
  __shared__ float se[TED_];
  int b = blockIdx.x, t = threadIdx.x;
  for (int i = t; i < TED_; i += 256) {
    float e = emb[(size_t)b * TED_ + i];
    se[i] = e / (1.f + __expf(-e));
  }
  __syncthreads();
  for (int n = t; n < 2 * D_; n += 256) {
    float acc = 0.f;
#pragma unroll 4
    for (int k2 = 0; k2 < TED_; ++k2)
      acc += se[k2] * W[(size_t)k2 * (2 * D_) + n];
    out[(size_t)b * (2 * D_) + n] = acc + bias[n];
  }
}

// ---------------- K8: LN2 + (1+scale)/shift modulation + SiLU -> f16 ----------------
__global__ void k_ln2mod(const float* __restrict__ y, const float* __restrict__ g,
                         const float* __restrict__ bb, const float* __restrict__ embo,
                         _Float16* __restrict__ h16) {
  int row = blockIdx.x;
  int t = threadIdx.x;
  int b = row / T_;
  __shared__ float red[16];
  float y0 = y[(size_t)row * D_ + t];
  float y1 = y[(size_t)row * D_ + t + 256];
  float s = y0 + y1, s2 = y0 * y0 + y1 * y1;
#pragma unroll
  for (int off = 16; off; off >>= 1) {
    s  += __shfl_xor(s, off, 32);
    s2 += __shfl_xor(s2, off, 32);
  }
  int w = t >> 5, lane = t & 31;
  if (lane == 0) { red[w] = s; red[8 + w] = s2; }
  __syncthreads();
  float ts = 0.f, ts2 = 0.f;
#pragma unroll
  for (int i = 0; i < 8; ++i) { ts += red[i]; ts2 += red[8 + i]; }
  float mean = ts * (1.f / D_);
  float var  = ts2 * (1.f / D_) - mean * mean;
  float inv  = rsqrtf(var + 1e-5f);
#pragma unroll
  for (int rep = 0; rep < 2; ++rep) {
    int d = t + rep * 256;
    float yy = rep ? y1 : y0;
    float sc = embo[(size_t)b * 1024 + d];
    float sh = embo[(size_t)b * 1024 + 512 + d];
    float vv = ((yy - mean) * inv * g[d] + bb[d]) * (1.f + sc) + sh;
    h16[(size_t)row * D_ + d] = (_Float16)(vv / (1.f + __expf(-vv)));
  }
}

// ---------------- launch ----------------
extern "C" void kernel_launch(void* const* d_in, const int* in_sizes, int n_in,
                              void* d_out, int out_size, void* d_ws, size_t ws_size,
                              hipStream_t stream) {
  (void)in_sizes; (void)n_in; (void)out_size; (void)ws_size;
  const float* x     = (const float*)d_in[0];
  const float* emb   = (const float*)d_in[1];
  const float* Wq    = (const float*)d_in[2];
  const float* Wk    = (const float*)d_in[3];
  const float* Wv    = (const float*)d_in[4];
  const float* Wsig  = (const float*)d_in[5];
  const float* ln1_g = (const float*)d_in[6];
  const float* ln1_b = (const float*)d_in[7];
  const float* emb_W = (const float*)d_in[8];
  const float* emb_b = (const float*)d_in[9];
  const float* ln2_g = (const float*)d_in[10];
  const float* ln2_b = (const float*)d_in[11];
  const float* out_W = (const float*)d_in[12];
  const float* out_b = (const float*)d_in[13];

  float* y_out   = (float*)d_out;                       // B,T,D
  float* series  = y_out + (size_t)BT_ * D_;            // B,H,T,T
  float* prior   = series + (size_t)B_ * H_ * T_ * T_;  // B,H,T,T
  float* sfull   = prior  + (size_t)B_ * H_ * T_ * T_;  // B,H,T,T

  char* p = (char*)d_ws;
  _Float16* xn16 = (_Float16*)p; p += (size_t)BT_ * D_ * 2;
  _Float16* q16  = (_Float16*)p; p += (size_t)BT_ * D_ * 2;
  _Float16* k16  = (_Float16*)p; p += (size_t)BT_ * D_ * 2;
  _Float16* vT16 = (_Float16*)p; p += (size_t)BT_ * D_ * 2;
  _Float16* h16  = (_Float16*)p; p += (size_t)BT_ * D_ * 2;
  _Float16* Wq16 = (_Float16*)p; p += (size_t)D_ * D_ * 2;
  _Float16* Wk16 = (_Float16*)p; p += (size_t)D_ * D_ * 2;
  _Float16* Wv16 = (_Float16*)p; p += (size_t)D_ * D_ * 2;
  _Float16* oW16 = (_Float16*)p; p += (size_t)D_ * D_ * 2;
  float* sig     = (float*)p;    p += (size_t)B_ * H_ * T_ * 4;
  float* embo    = (float*)p;    p += (size_t)B_ * 1024 * 4;
  float* ymid    = (float*)p;    p += (size_t)BT_ * D_ * 4;

  // 1) LN1 + sigma
  k_ln1_sigma<<<BT_, 256, 0, stream>>>(x, ln1_g, ln1_b, Wsig, xn16, sig);
  // 2) weight converts
  k_cvt16<<<(D_ * D_) / 256, 256, 0, stream>>>(Wq, Wq16, D_ * D_);
  k_cvt16<<<(D_ * D_) / 256, 256, 0, stream>>>(Wk, Wk16, D_ * D_);
  k_cvt16<<<(D_ * D_) / 256, 256, 0, stream>>>(Wv, Wv16, D_ * D_);
  k_cvt16<<<(D_ * D_) / 256, 256, 0, stream>>>(out_W, oW16, D_ * D_);
  // 3) prior + sigma_full (streaming, non-temporal)
  k_prior<<<(B_ * H_ * T_ * T_) / 4 / 256, 256, 0, stream>>>(sig, prior, sfull);
  // 4) Q, K, V^T projections (block tile 128x128)
  dim3 gg(BT_ / 128, D_ / 128);
  k_gemm<<<gg, 256, 0, stream>>>(xn16, Wq16, q16, nullptr, nullptr, nullptr, 0);
  k_gemm<<<gg, 256, 0, stream>>>(xn16, Wk16, k16, nullptr, nullptr, nullptr, 0);
  k_gemm<<<gg, 256, 0, stream>>>(xn16, Wv16, vT16, nullptr, nullptr, nullptr, 2);
  // 5) scores + softmax -> series (d_out)
  k_attn_scores<<<dim3(T_ / 16, B_ * H_), 256, 0, stream>>>(q16, k16, series);
  // 6) y = series @ v
  k_yv<<<dim3(T_ / 128, B_ * H_), 256, 0, stream>>>(series, vT16, ymid);
  // 7) stylization emb GEMM
  k_emb<<<B_, 256, 0, stream>>>(emb, emb_W, emb_b, embo);
  // 8) LN2 + modulation + SiLU
  k_ln2mod<<<BT_, 256, 0, stream>>>(ymid, ln2_g, ln2_b, embo, h16);
  // 9) out projection + bias + residual -> y_out (non-temporal)
  k_gemm<<<gg, 256, 0, stream>>>(h16, oW16, nullptr, y_out, x, out_b, 3);
}
